// Model_39676907881612
// MI455X (gfx1250) — matrix operand order repaired and checked by reading, not compiled, not run on hardware
//
#include <hip/hip_runtime.h>

typedef __attribute__((ext_vector_type(16))) _Float16 v16h;
typedef __attribute__((ext_vector_type(8)))  _Float16 v8h;
typedef __attribute__((ext_vector_type(4)))  _Float16 v4h;
typedef __attribute__((ext_vector_type(2)))  _Float16 v2h;
typedef __attribute__((ext_vector_type(8)))  float    v8f;
typedef __attribute__((ext_vector_type(4)))  float    v4f;
typedef __attribute__((ext_vector_type(2)))  float    v2f;

#define B_      16
#define SQ_     4096
#define SKV_    4096
#define D_      64
#define KV_TILE 64
#define NT      (SKV_ / KV_TILE)   // 64 iterations
#define WAVES   8
#define QPW     32                 // q rows per wave (two 16-row A tiles)
#define Q_TILE  (WAVES * QPW)      // 256 q rows per block

// LDS row stride in halves: 72 halves = 144B, multiple of 16B (b128-aligned),
// and 36-bank stagger between rows (conflict-free b128 reads across 16 lanes).
#define KS 72   // K tile  [64][KS]  row-major [kv][d]
#define VS 72   // V tile  [64][VS]  transposed [d][kv]
#define PS 72   // P buffer [16][PS] per wave per q-subtile

union AFrag { v16h v; v8h h[2]; };

__device__ __forceinline__ float swz_max(float x, float y) { return fmaxf(x, y); }
__device__ __forceinline__ float fast_exp2(float x) { return __builtin_amdgcn_exp2f(x); }

__global__ __launch_bounds__(256)
void fattn_f16_wmma(const float* __restrict__ q, const float* __restrict__ k,
                    const float* __restrict__ v, const float* __restrict__ scale_p,
                    float* __restrict__ out)
{
    __shared__ _Float16 lds_k[KV_TILE * KS];                 //  9216 B
    __shared__ _Float16 lds_vt[D_ * VS];                     //  9216 B
    __shared__ _Float16 lds_p[WAVES * 2 * 16 * PS];          // 36864 B

    const int tid  = threadIdx.x;
    const int lane = tid & 31;
    const int wave = tid >> 5;
    const int m    = lane & 15;   // A row / B column index
    const int g    = lane >> 4;   // half-wave group
    const int bq   = blockIdx.x;  // q-tile (256 rows)
    const int b    = blockIdx.y;  // batch

    // log2-domain softmax: fold log2(e) into the score scale so every exp in the
    // hot loop is a bare v_exp_f32 (HW exp2) with no pre-multiply.
    const float scale2 = scale_p[0] * 1.44269504088896340736f;

    // ---- Q fragments for two 16-row subtiles (scale folded in, f32 -> f16) ----
    AFrag qf[2][2];               // [q-subtile u][d-chunk c]
#pragma unroll
    for (int u = 0; u < 2; ++u) {
        const int qrow = bq * Q_TILE + wave * QPW + u * 16 + m;
        const float* qptr = q + ((size_t)b * SQ_ + qrow) * D_;
#pragma unroll
        for (int c = 0; c < 2; ++c)
#pragma unroll
            for (int h = 0; h < 2; ++h) {
                const int d0 = c * 32 + g * 16 + h * 8;   // A: K = g*16 + h*8 .. +8
                v4f f0 = *(const v4f*)(qptr + d0);
                v4f f1 = *(const v4f*)(qptr + d0 + 4);
                v8h hv;
#pragma unroll
                for (int j = 0; j < 4; ++j) {
                    hv[j]     = (_Float16)(f0[j] * scale2);
                    hv[4 + j] = (_Float16)(f1[j] * scale2);
                }
                qf[u][c].h[h] = hv;
            }
    }

    // all-ones B fragment for row-sum WMMA
    v16h ones;
#pragma unroll
    for (int i = 0; i < 16; ++i) ones[i] = (_Float16)1.0f;

    // ---- accumulators + online-softmax state (rows M = i + 8*g) ----
    v8f acc[2][4];
#pragma unroll
    for (int u = 0; u < 2; ++u)
#pragma unroll
        for (int j = 0; j < 4; ++j)
#pragma unroll
            for (int i = 0; i < 8; ++i) acc[u][j][i] = 0.0f;
    float mrow[2][8], lrow[2][8];
#pragma unroll
    for (int u = 0; u < 2; ++u)
#pragma unroll
        for (int i = 0; i < 8; ++i) { mrow[u][i] = -1e30f; lrow[u][i] = 0.0f; }

    _Float16* pbuf0 = &lds_p[(wave * 2 + 0) * 16 * PS];
    _Float16* pbuf1 = &lds_p[(wave * 2 + 1) * 16 * PS];
    const float* kbase = k + (size_t)b * SKV_ * D_;
    const float* vbase = v + (size_t)b * SKV_ * D_;

    // warm the cache for tile 0 (global_prefetch_b8; 256 threads x 64B = 16KB tile)
    __builtin_prefetch(kbase + tid * 16, 0, 3);
    __builtin_prefetch(vbase + tid * 16, 0, 3);

    for (int t = 0; t < NT; ++t) {
        const float* kt = kbase + (size_t)t * KV_TILE * D_;
        const float* vt = vbase + (size_t)t * KV_TILE * D_;
        __syncthreads();   // previous tile fully consumed
        // ---- K: global -> LDS row-major (f32->f16, packed converts) ----
#pragma unroll
        for (int it = 0; it < 4; ++it) {
            const int idx = tid + it * 256;       // 1024 float4 chunks per tile
            const int row = idx >> 4;             // kv
            const int c4  = (idx & 15) * 4;       // d
            v4f fk = *(const v4f*)(kt + row * D_ + c4);
            v4h hk;
#pragma unroll
            for (int j = 0; j < 4; ++j) hk[j] = (_Float16)fk[j];
            *(v4h*)(&lds_k[row * KS + c4]) = hk;
        }
        // ---- V: 2x2 micro-tiles -> transposed LDS with packed b32 stores ----
#pragma unroll
        for (int it = 0; it < 4; ++it) {
            const int idx = tid + it * 256;       // 1024 2x2 micro-tiles
            const int kv2 = (idx >> 5) * 2;       // kv pair
            const int d2  = (idx & 31) * 2;       // d pair
            v2f a0 = *(const v2f*)(vt + kv2 * D_ + d2);
            v2f a1 = *(const v2f*)(vt + (kv2 + 1) * D_ + d2);
            v2h p0; p0[0] = (_Float16)a0[0]; p0[1] = (_Float16)a1[0];
            v2h p1; p1[0] = (_Float16)a0[1]; p1[1] = (_Float16)a1[1];
            *(v2h*)(&lds_vt[d2 * VS + kv2])       = p0;
            *(v2h*)(&lds_vt[(d2 + 1) * VS + kv2]) = p1;
        }
        // ---- prefetch tile t+1 (overlaps with compute below) ----
        if (t + 1 < NT) {
            __builtin_prefetch(kt + (size_t)KV_TILE * D_ + tid * 16, 0, 3);
            __builtin_prefetch(vt + (size_t)KV_TILE * D_ + tid * 16, 0, 3);
        }
        __syncthreads();   // tile visible to all waves

        // ---- S = (Q*scale*log2e) @ K^T : each K B-fragment feeds both q-subtiles ----
        v8f s[2][4];
#pragma unroll
        for (int u = 0; u < 2; ++u)
#pragma unroll
            for (int n = 0; n < 4; ++n)
#pragma unroll
                for (int i = 0; i < 8; ++i) s[u][n][i] = 0.0f;
#pragma unroll
        for (int n = 0; n < 4; ++n)
#pragma unroll
            for (int c = 0; c < 2; ++c) {
                AFrag bf;   // B: row(kv)=n*16+m, K(d)=c*32 + g*16 .. +16 contiguous
                const _Float16* bp = &lds_k[(n * 16 + m) * KS + c * 32 + g * 16];
                bf.h[0] = *(const v8h*)(bp);
                bf.h[1] = *(const v8h*)(bp + 8);
                s[0][n] = __builtin_amdgcn_wmma_f32_16x16x32_f16(false, qf[0][c].v, false, bf.v,
                                                                 (short)0, s[0][n], false, false);
                s[1][n] = __builtin_amdgcn_wmma_f32_16x16x32_f16(false, qf[1][c].v, false, bf.v,
                                                                 (short)0, s[1][n], false, false);
            }

        // ---- per-subtile online softmax in log2 domain (P -> LDS) ----
#pragma unroll
        for (int u = 0; u < 2; ++u) {
            _Float16* pbuf = (u == 0) ? pbuf0 : pbuf1;
            float mcur[8];
#pragma unroll
            for (int i = 0; i < 8; ++i)
                mcur[i] = fmaxf(fmaxf(s[u][0][i], s[u][1][i]), fmaxf(s[u][2][i], s[u][3][i]));
#pragma unroll
            for (int i = 0; i < 8; ++i) {
                float x = mcur[i];
                x = swz_max(x, __int_as_float(__builtin_amdgcn_ds_swizzle(__float_as_int(x), 0x03F)));
                x = swz_max(x, __int_as_float(__builtin_amdgcn_ds_swizzle(__float_as_int(x), 0x05F)));
                x = swz_max(x, __int_as_float(__builtin_amdgcn_ds_swizzle(__float_as_int(x), 0x09F)));
                x = swz_max(x, __int_as_float(__builtin_amdgcn_ds_swizzle(__float_as_int(x), 0x11F)));
                mcur[i] = x;
            }
#pragma unroll
            for (int i = 0; i < 8; ++i) {
                const float mn    = fmaxf(mrow[u][i], mcur[i]);
                const float alpha = fast_exp2(mrow[u][i] - mn);
                mrow[u][i] = mn;
                lrow[u][i] *= alpha;               // psum added after PV phase
#pragma unroll
                for (int j = 0; j < 4; ++j) acc[u][j][i] *= alpha;
            }
#pragma unroll
            for (int n = 0; n < 4; ++n)
#pragma unroll
                for (int i = 0; i < 8; ++i) {
                    const float p = fast_exp2(s[u][n][i] - mrow[u][i]);
                    pbuf[(g * 8 + i) * PS + n * 16 + m] = (_Float16)p;
                }
        }

        // ---- PV phase: pf/psum live only here (S registers are dead) ----
        v8f psum[2];
#pragma unroll
        for (int u = 0; u < 2; ++u)
#pragma unroll
            for (int i = 0; i < 8; ++i) psum[u][i] = 0.0f;
#pragma unroll
        for (int c = 0; c < 2; ++c) {
            AFrag pf[2];
#pragma unroll
            for (int u = 0; u < 2; ++u) {
                _Float16* pbuf = (u == 0) ? pbuf0 : pbuf1;
                pf[u].h[0] = *(const v8h*)(&pbuf[m * PS + c * 32 + g * 16]);
                pf[u].h[1] = *(const v8h*)(&pbuf[m * PS + c * 32 + g * 16 + 8]);
            }
            // row sums of P via WMMA against all-ones B
            psum[0] = __builtin_amdgcn_wmma_f32_16x16x32_f16(false, pf[0].v, false, ones,
                                                             (short)0, psum[0], false, false);
            psum[1] = __builtin_amdgcn_wmma_f32_16x16x32_f16(false, pf[1].v, false, ones,
                                                             (short)0, psum[1], false, false);
            // O += P @ V : each V B-fragment feeds both q-subtiles
#pragma unroll
            for (int j = 0; j < 4; ++j) {
                AFrag vf;   // B: row(d)=j*16+m, K(kv)=c*32 + g*16 .. +16 contiguous
                const _Float16* vp = &lds_vt[(j * 16 + m) * VS + c * 32 + g * 16];
                vf.h[0] = *(const v8h*)(vp);
                vf.h[1] = *(const v8h*)(vp + 8);
                acc[0][j] = __builtin_amdgcn_wmma_f32_16x16x32_f16(false, pf[0].v, false, vf.v,
                                                                   (short)0, acc[0][j], false, false);
                acc[1][j] = __builtin_amdgcn_wmma_f32_16x16x32_f16(false, pf[1].v, false, vf.v,
                                                                   (short)0, acc[1][j], false, false);
            }
        }
#pragma unroll
        for (int u = 0; u < 2; ++u)
#pragma unroll
            for (int i = 0; i < 8; ++i) lrow[u][i] += psum[u][i];
    }

    // ---- epilogue: normalize by row sum, store fp32 ----
#pragma unroll
    for (int u = 0; u < 2; ++u) {
#pragma unroll
        for (int i = 0; i < 8; ++i) lrow[u][i] = 1.0f / lrow[u][i];
        float* optr = out + ((size_t)b * SQ_ + (size_t)bq * Q_TILE + wave * QPW + u * 16) * D_;
#pragma unroll
        for (int j = 0; j < 4; ++j)
#pragma unroll
            for (int i = 0; i < 8; ++i)
                optr[(g * 8 + i) * D_ + j * 16 + m] = acc[u][j][i] * lrow[u][i];
    }
}

extern "C" void kernel_launch(void* const* d_in, const int* in_sizes, int n_in,
                              void* d_out, int out_size, void* d_ws, size_t ws_size,
                              hipStream_t stream) {
    (void)in_sizes; (void)n_in; (void)out_size; (void)d_ws; (void)ws_size;
    const float* q = (const float*)d_in[0];
    const float* k = (const float*)d_in[1];
    const float* v = (const float*)d_in[2];
    const float* s = (const float*)d_in[3];
    float* o = (float*)d_out;
    dim3 grid(SQ_ / Q_TILE, B_);
    dim3 block(256);
    fattn_f16_wmma<<<grid, block, 0, stream>>>(q, k, v, s, o);
}
